// Multi_EfficientAttention_34256659153603
// MI455X (gfx1250) — matrix operand order repaired and checked by reading, not compiled, not run on hardware
//
#include <hip/hip_runtime.h>
#include <hip/hip_bf16.h>

// ---------------------------------------------------------------------------
// Problem constants (from reference): B=4, NX=NY=16384, CX=CY=CK=CV=256, H=8
// ---------------------------------------------------------------------------
#define NB   4
#define NTOK 16384
#define NC   256
#define NH   8
#define ND   32           // per-head dim (dk = dv = 32)
#define MTOT (NB * NTOK)  // 65536 rows total

typedef __attribute__((ext_vector_type(16))) _Float16 v16h;
typedef __attribute__((ext_vector_type(8)))  _Float16 v8h;
typedef __attribute__((ext_vector_type(8)))  float    v8f;
typedef __attribute__((ext_vector_type(4)))  unsigned v4u;
typedef __attribute__((ext_vector_type(8)))  int      v8i;
typedef __attribute__((ext_vector_type(4)))  int      v4i;

union HFrag { v16h v; v8h h[2]; };

// ---- WMMA fragment loaders (wave32, 16-bit A/B layouts per CDNA5 ISA 7.12.2)
// A (16x32, MxK): lane m = l&15; lanes<16 hold K {0..7,16..23}, lanes>=16 hold
// K {8..15,24..31} as two contiguous 8-half runs.
__device__ __forceinline__ v16h load_a_frag(const _Float16* lds, int stride, int lane) {
    const int m  = lane & 15;
    const int hi = (lane >> 4) << 3;   // 0 or 8
    const _Float16* r = lds + m * stride;
    HFrag f;
    f.h[0] = *(const v8h*)(r + hi);        // K = hi .. hi+7
    f.h[1] = *(const v8h*)(r + 16 + hi);   // K = 16+hi .. 23+hi
    return f.v;
}

// B (32x16, KxN) from LDS tile laid out [n][k] with row stride `stride`:
// lane n = l&15; lanes<16 hold K {0..7,16..23}, lanes>=16 hold K {8..15,24..31}.
__device__ __forceinline__ v16h load_b_frag(const _Float16* lds, int stride, int lane) {
    const int n  = lane & 15;
    const int k0 = (lane >> 4) << 3;   // 0 or 8
    const _Float16* r = lds + n * stride + k0;
    HFrag f;
    f.h[0] = *(const v8h*)(r);
    f.h[1] = *(const v8h*)(r + 16);
    return f.v;
}

__device__ __forceinline__ v8f wmma_f16(v16h a, v16h b, v8f c) {
    return __builtin_amdgcn_wmma_f32_16x16x32_f16(false, a, false, b, (short)0, c, false, false);
}

// ---------------------------------------------------------------------------
// Tensor Data Mover: 1-D contiguous f16 DMA, global -> LDS.
// D# built per CDNA5 ISA (group0/group1 bitfields):
//   group0: [1:0]=count=1, [63:32]=lds_addr, [120:64]=global_addr, [127:126]=2
//   group1: [17:16]=data_size(1 => 2B), tensor_dim0=N, tensor_dim1=1,
//           tile_dim0=N (<=65535), tile_dim1=1, stride0=N
// nHalf must be <= 65535.
// ---------------------------------------------------------------------------
__device__ __forceinline__ void tdm_load_1d(unsigned ldsOff, const void* gsrc, unsigned nHalf) {
    const unsigned long long ga = (unsigned long long)(uintptr_t)gsrc;
    v4u g0;
    g0.x = 1u;                                            // count=1 (user D#)
    g0.y = ldsOff;                                        // lds_addr (bytes)
    g0.z = (unsigned)(ga & 0xFFFFFFFFu);                  // global_addr[31:0]
    g0.w = (unsigned)((ga >> 32) & 0x01FFFFFFu) | (2u << 30);  // addr[56:32] | type=2
    v8i g1;
    g1[0] = (int)(1u << 16);                              // data_size=1 (2B), mask=0
    g1[1] = (int)((nHalf & 0xFFFFu) << 16);               // tensor_dim0[15:0]
    g1[2] = (int)(((nHalf >> 16) & 0xFFFFu) | (1u << 16));// tensor_dim0[31:16], tensor_dim1=1
    g1[3] = (int)((nHalf & 0xFFFFu) << 16);               // tile_dim0 = N
    g1[4] = 1;                                            // tile_dim1 = 1, tile_dim2 = 0
    g1[5] = (int)nHalf;                                   // tensor_dim0_stride[31:0]
    g1[6] = 0;
    g1[7] = 0;
    v4i gz = {0, 0, 0, 0};
#if __has_include(<hip/amd_detail/amd_gfx1250_TDM.h>)
    v8i gz8 = {0, 0, 0, 0, 0, 0, 0, 0};
    __builtin_amdgcn_tensor_load_to_lds(g0, g1, gz, gz, gz8, 0);   // 6-arg toolchain
#else
    __builtin_amdgcn_tensor_load_to_lds(g0, g1, gz, gz, 0);        // 5-arg toolchain
#endif
}

// Generic-pointer -> LDS byte offset (LDS offset is the low 32 bits of the address)
__device__ __forceinline__ unsigned lds_offset(const void* p) {
    return (unsigned)(uintptr_t)p;
}

// Order-preserving float <-> uint mapping for atomicMax on floats
__device__ __forceinline__ unsigned orderKey(float f) {
    unsigned u = __float_as_uint(f);
    return (u & 0x80000000u) ? ~u : (u | 0x80000000u);
}
__device__ __forceinline__ float invKey(unsigned k) {
    return (k & 0x80000000u) ? __uint_as_float(k & 0x7fffffffu) : __uint_as_float(~k);
}

// ---------------------------------------------------------------------------
// K0a: zero the small accumulator region (colmax/colsum/context)
// ---------------------------------------------------------------------------
__global__ __launch_bounds__(256) void ws_init(unsigned* p, int n) {
    int i = blockIdx.x * 256 + threadIdx.x;
    if (i < n) p[i] = 0u;
}

// K0b: convert a weight matrix f32 -> f16 (contiguous), n4 = n/4 float4s
__global__ __launch_bounds__(256) void f32_to_f16(const float* __restrict__ src,
                                                  _Float16* __restrict__ dst, int n4) {
    int i = blockIdx.x * 256 + threadIdx.x;
    if (i < n4) {
        float4 v = ((const float4*)src)[i];
        int o = i * 4;
        dst[o + 0] = (_Float16)v.x; dst[o + 1] = (_Float16)v.y;
        dst[o + 2] = (_Float16)v.z; dst[o + 3] = (_Float16)v.w;
    }
}

// ---------------------------------------------------------------------------
// K1: row-major GEMM  out[m, c] = sum_k in[m,k] * Wh[c,k] + bias[c]
//     Wh is pre-converted f16 (contiguous rows) -> W tile staged by TDM.
//     grid = (Mrows/64, 2), block = 256 (8 waves).
//     Block tile 64x128; wave (mw = w&3, nw = w>>2) owns 16 rows x 64 cols
//     (4 accumulators), 32 WMMAs per wave over K=256.
// ---------------------------------------------------------------------------
template<bool ADD_POS, bool IN_F16, bool OUT_F32>
__global__ __launch_bounds__(256) void proj_gemm(
    const void* __restrict__ in_, const float* __restrict__ pos,
    const _Float16* __restrict__ Wh, const float* __restrict__ bias,
    void* __restrict__ out_)
{
    __shared__ _Float16 sA[64 * 256];    // 32 KB activation tile (f16)
    __shared__ _Float16 sW[128 * 256];   // 64 KB weight tile (f16)

    const int tid  = threadIdx.x;
    const int lane = tid & 31;
    const int wave = tid >> 5;
    const int row0 = blockIdx.x * 64;
    const int col0 = blockIdx.y * 128;

    // ---- wave 0: kick off TDM DMA of the weight tile (128*256 = 32768 halfs)
    if (wave == 0) {
        tdm_load_1d(lds_offset(sW), Wh + (size_t)col0 * 256, 128 * 256);
        if constexpr (IN_F16)   // activation tile already f16 & contiguous
            tdm_load_1d(lds_offset(sA), (const _Float16*)in_ + (size_t)row0 * 256, 64 * 256);
    }

    // ---- stage activations (f32 path): 64 rows x 256 -> f16, overlaps the DMA
    if constexpr (!IN_F16) {
        const float* in = (const float*)in_;
        const float4* src = (const float4*)(in + (size_t)row0 * 256);
        const float4* ps  = ADD_POS ? (const float4*)(pos + (size_t)row0 * 256) : nullptr;
        __builtin_prefetch((const void*)src, 0, 3);
        for (int i = tid; i < 4096; i += 256) {       // 4096 float4 = 16384 f32
            float4 a = src[i];
            if constexpr (ADD_POS) {
                float4 p = ps[i];
                a.x += p.x; a.y += p.y; a.z += p.z; a.w += p.w;
            }
            int o = i * 4;
            sA[o + 0] = (_Float16)a.x; sA[o + 1] = (_Float16)a.y;
            sA[o + 2] = (_Float16)a.z; sA[o + 3] = (_Float16)a.w;
        }
    }

    if (wave == 0) __builtin_amdgcn_s_wait_tensorcnt(0);
    __syncthreads();

    const int mw = wave & 3;         // M tile 0..3
    const int nw = wave >> 2;        // N group 0..1 (64 cols each)
    const _Float16* aBase = sA + (size_t)(mw * 16) * 256;
    const _Float16* bBase = sW + (size_t)(nw * 64) * 256;

    v8f acc0 = {}, acc1 = {}, acc2 = {}, acc3 = {};
#pragma unroll
    for (int kb = 0; kb < 256; kb += 32) {
        v16h a  = load_a_frag(aBase + kb, 256, lane);
        v16h b0 = load_b_frag(bBase + kb,            256, lane);
        v16h b1 = load_b_frag(bBase + 16 * 256 + kb, 256, lane);
        v16h b2 = load_b_frag(bBase + 32 * 256 + kb, 256, lane);
        v16h b3 = load_b_frag(bBase + 48 * 256 + kb, 256, lane);
        acc0 = wmma_f16(a, b0, acc0);
        acc1 = wmma_f16(a, b1, acc1);
        acc2 = wmma_f16(a, b2, acc2);
        acc3 = wmma_f16(a, b3, acc3);
    }

    const int n     = lane & 15;
    const int mBase = (lane >> 4) * 8;
    v8f accs[4] = {acc0, acc1, acc2, acc3};
#pragma unroll
    for (int nt = 0; nt < 4; ++nt) {
        const int col  = col0 + nw * 64 + nt * 16 + n;
        const float bv = bias[col];
#pragma unroll
        for (int r = 0; r < 8; ++r) {
            const int row = row0 + mw * 16 + mBase + r;
            const float val = accs[nt][r] + bv;
            if constexpr (OUT_F32)
                ((float*)out_)[(size_t)row * 256 + col] = val;
            else
                ((_Float16*)out_)[(size_t)row * 256 + col] = (_Float16)val;
        }
    }
}

// ---------------------------------------------------------------------------
// K2/K3: per-(batch, channel) softmax stats over 16384 tokens of k-logits
// ---------------------------------------------------------------------------
__global__ __launch_bounds__(256) void kv_colmax(const _Float16* __restrict__ klog,
                                                 unsigned* __restrict__ colmaxU) {
    const int bb = blockIdx.y;
    const int c  = threadIdx.x;                 // channel 0..255
    const int r0 = blockIdx.x * 512;
    const _Float16* p = klog + ((size_t)bb * NTOK + r0) * NC + c;
    float m = -3.0e38f;
#pragma unroll 8
    for (int i = 0; i < 512; ++i)
        m = fmaxf(m, (float)p[(size_t)i * NC]);
    atomicMax(&colmaxU[bb * NC + c], orderKey(m));
}

__global__ __launch_bounds__(256) void kv_colsum(const _Float16* __restrict__ klog,
                                                 const unsigned* __restrict__ colmaxU,
                                                 float* __restrict__ colsum) {
    const int bb = blockIdx.y;
    const int c  = threadIdx.x;
    const int r0 = blockIdx.x * 512;
    const float cm = invKey(colmaxU[bb * NC + c]);
    const _Float16* p = klog + ((size_t)bb * NTOK + r0) * NC + c;
    float s = 0.f;
#pragma unroll 8
    for (int i = 0; i < 512; ++i)
        s += expf((float)p[(size_t)i * NC] - cm);
    atomicAdd(&colsum[bb * NC + c], s);
}

// ---------------------------------------------------------------------------
// K4: context[b,h,c,cv] += sum_n exp(k[n,c]-max_c) * v[n,cv]
//     grid = (16 token-chunks, B); block = 8 waves; wave w == head w.
//     Each wave stages 32-token x 32-channel transposed tiles in its own LDS
//     slice (row stride 40 halfs: 16B-aligned, bank-spread) and runs 4 WMMA.
// ---------------------------------------------------------------------------
__global__ __launch_bounds__(256) void kv_context(const _Float16* __restrict__ klog,
                                                  const _Float16* __restrict__ vbuf,
                                                  const unsigned* __restrict__ colmaxU,
                                                  float* __restrict__ context) {
    __shared__ _Float16 sEkT[NH][32 * 40];   // [head][channel][token]
    __shared__ _Float16 sVT [NH][32 * 40];   // [head][v-channel][token]

    const int bb   = blockIdx.y;
    const int tok0 = blockIdx.x * 1024;
    const int wave = threadIdx.x >> 5;
    const int lane = threadIdx.x & 31;
    const int h    = wave;
    const int c0   = h * ND;
    const float cm = invKey(colmaxU[bb * NC + c0 + lane]);

    const _Float16* kb = klog + (size_t)bb * NTOK * NC;
    const _Float16* vb = vbuf + (size_t)bb * NTOK * NC;

    v8f a00 = {}, a01 = {}, a10 = {}, a11 = {};

    for (int step = 0; step < 32; ++step) {
        const int t0 = tok0 + step * 32;
        // lane owns one channel column; transpose 32 tokens into LDS
#pragma unroll 4
        for (int i = 0; i < 32; ++i) {
            const size_t off = (size_t)(t0 + i) * NC + c0 + lane;
            sEkT[h][lane * 40 + i] = (_Float16)expf((float)kb[off] - cm);
            sVT [h][lane * 40 + i] = vb[off];
        }
        // wave-private LDS slice; wave32 lockstep + in-order DS => no barrier
        v16h fa0 = load_a_frag(&sEkT[h][0],       40, lane);  // channels c0..c0+15
        v16h fa1 = load_a_frag(&sEkT[h][16 * 40], 40, lane);  // channels c0+16..31
        v16h fb0 = load_b_frag(&sVT [h][0],       40, lane);  // cv 0..15
        v16h fb1 = load_b_frag(&sVT [h][16 * 40], 40, lane);  // cv 16..31
        a00 = wmma_f16(fa0, fb0, a00);
        a01 = wmma_f16(fa0, fb1, a01);
        a10 = wmma_f16(fa1, fb0, a10);
        a11 = wmma_f16(fa1, fb1, a11);
    }

    float* ctx = context + (size_t)(bb * NH + h) * ND * ND;
    const int n  = lane & 15;
    const int mb = (lane >> 4) * 8;
#pragma unroll
    for (int r = 0; r < 8; ++r) {
        atomicAdd(&ctx[(mb + r) * ND + n],            a00[r]);
        atomicAdd(&ctx[(mb + r) * ND + 16 + n],       a01[r]);
        atomicAdd(&ctx[(16 + mb + r) * ND + n],       a10[r]);
        atomicAdd(&ctx[(16 + mb + r) * ND + 16 + n],  a11[r]);
    }
}

// ---------------------------------------------------------------------------
// K5: attended[row, h*32+cv] = sum_c softmax_c(q[row, h*32+c]) * ctx[h][c,cv]
//     grid = MTOT/16; block = 8 waves; wave w == head w (16x32 WMMA output).
// ---------------------------------------------------------------------------
__global__ __launch_bounds__(256) void attn_apply(const _Float16* __restrict__ qlog,
                                                  const float* __restrict__ context,
                                                  const float* __restrict__ colsum,
                                                  _Float16* __restrict__ att) {
    __shared__ _Float16 sQ[16 * 256];        // softmaxed q tile
    __shared__ _Float16 sCtxT[NH][32 * 40];  // [head][cv][c], 1/colsum folded in

    const int tid  = threadIdx.x;
    const int lane = tid & 31;
    const int wave = tid >> 5;
    const int row0 = blockIdx.x * 16;
    const int bb   = row0 / NTOK;

    // 1) q softmax per (row, head) over 32 channels
    if (tid < 128) {
        const int r = tid >> 3, hh = tid & 7;
        const _Float16* qp = qlog + (size_t)(row0 + r) * NC + hh * ND;
        float vals[ND];
        float mx = -3.0e38f;
#pragma unroll
        for (int i = 0; i < ND; ++i) { vals[i] = (float)qp[i]; mx = fmaxf(mx, vals[i]); }
        float s = 0.f;
#pragma unroll
        for (int i = 0; i < ND; ++i) { vals[i] = expf(vals[i] - mx); s += vals[i]; }
        const float inv = 1.f / s;
#pragma unroll
        for (int i = 0; i < ND; ++i)
            sQ[r * 256 + hh * ND + i] = (_Float16)(vals[i] * inv);
    }

    // 2) stage transposed, colsum-normalized context: 8 heads x 32x32
    for (int g = tid; g < NH * ND * ND; g += 256) {
        const int hh = g >> 10, rem = g & 1023, c = rem >> 5, cv = rem & 31;
        const float val = context[((size_t)(bb * NH + hh) * ND + c) * ND + cv]
                        * (1.0f / colsum[bb * NC + hh * ND + c]);
        sCtxT[hh][cv * 40 + c] = (_Float16)val;
    }
    __syncthreads();

    const int h = wave;
    v8f acc0 = {}, acc1 = {};
    v16h fa  = load_a_frag(sQ + h * ND, 256, lane);        // K = head channels
    v16h fb0 = load_b_frag(&sCtxT[h][0],       40, lane);  // cv 0..15
    v16h fb1 = load_b_frag(&sCtxT[h][16 * 40], 40, lane);  // cv 16..31
    acc0 = wmma_f16(fa, fb0, acc0);
    acc1 = wmma_f16(fa, fb1, acc1);

    const int n  = lane & 15;
    const int mb = (lane >> 4) * 8;
#pragma unroll
    for (int r = 0; r < 8; ++r) {
        const int row = row0 + mb + r;
        att[(size_t)row * 256 + h * ND + n]      = (_Float16)acc0[r];
        att[(size_t)row * 256 + h * ND + 16 + n] = (_Float16)acc1[r];
    }
}

// ---------------------------------------------------------------------------
// Host-side launcher
// ---------------------------------------------------------------------------
extern "C" void kernel_launch(void* const* d_in, const int* in_sizes, int n_in,
                              void* d_out, int out_size, void* d_ws, size_t ws_size,
                              hipStream_t stream) {
    (void)in_sizes; (void)n_in; (void)out_size; (void)ws_size;

    const float* x   = (const float*)d_in[0];
    const float* y   = (const float*)d_in[1];
    const float* xpe = (const float*)d_in[2];
    const float* ype = (const float*)d_in[3];
    const float* Wq  = (const float*)d_in[4];
    const float* bq  = (const float*)d_in[5];
    const float* Wk  = (const float*)d_in[6];
    const float* bk  = (const float*)d_in[7];
    const float* Wv  = (const float*)d_in[8];
    const float* bv  = (const float*)d_in[9];
    const float* Wr  = (const float*)d_in[10];
    const float* br  = (const float*)d_in[11];
    float* out = (float*)d_out;

    // workspace layout (f16 logits, f32 stats, f16 weights)
    char* ws = (char*)d_ws;
    const size_t SZ_H = (size_t)MTOT * NC * sizeof(_Float16);  // 32 MiB
    _Float16* qlog = (_Float16*)(ws);
    _Float16* klog = (_Float16*)(ws + SZ_H);
    _Float16* vbuf = (_Float16*)(ws + 2 * SZ_H);
    _Float16* att  = (_Float16*)(ws + 3 * SZ_H);
    char* stats = ws + 4 * SZ_H;
    unsigned* colmaxU = (unsigned*)(stats);
    float*    colsum  = (float*)(stats + 4096);
    float*    context = (float*)(stats + 8192);
    const int statDwords = NB * NC * 2 + NB * NH * ND * ND;    // 34816
    const size_t SZ_W = (size_t)NC * NC * sizeof(_Float16);    // 128 KiB
    _Float16* WqH = (_Float16*)(stats + 139264);
    _Float16* WkH = (_Float16*)((char*)WqH + SZ_W);
    _Float16* WvH = (_Float16*)((char*)WkH + SZ_W);
    _Float16* WrH = (_Float16*)((char*)WvH + SZ_W);

    dim3 blk(256);
    ws_init<<<dim3((statDwords + 255) / 256), blk, 0, stream>>>(colmaxU, statDwords);

    const int w4 = NC * NC / 4;  // 16384 float4 per weight matrix
    f32_to_f16<<<dim3(w4 / 256), blk, 0, stream>>>(Wq, WqH, w4);
    f32_to_f16<<<dim3(w4 / 256), blk, 0, stream>>>(Wk, WkH, w4);
    f32_to_f16<<<dim3(w4 / 256), blk, 0, stream>>>(Wv, WvH, w4);
    f32_to_f16<<<dim3(w4 / 256), blk, 0, stream>>>(Wr, WrH, w4);

    dim3 gGemm(MTOT / 64, 2);
    proj_gemm<true,  false, false><<<gGemm, blk, 0, stream>>>(x, xpe, WqH, bq, qlog);
    proj_gemm<true,  false, false><<<gGemm, blk, 0, stream>>>(y, ype, WkH, bk, klog);
    proj_gemm<false, false, false><<<gGemm, blk, 0, stream>>>(y, nullptr, WvH, bv, vbuf);

    kv_colmax<<<dim3(32, NB), blk, 0, stream>>>(klog, colmaxU);
    kv_colsum<<<dim3(32, NB), blk, 0, stream>>>(klog, colmaxU, colsum);

    kv_context<<<dim3(16, NB), blk, 0, stream>>>(klog, vbuf, colmaxU, context);

    attn_apply<<<dim3(MTOT / 16), blk, 0, stream>>>(qlog, context, colsum, att);

    proj_gemm<false, true, true><<<gGemm, blk, 0, stream>>>(att, nullptr, WrH, br, out);
}
